// Block_36575941492917
// MI455X (gfx1250) — matrix operand order repaired and checked by reading, not compiled, not run on hardware
//
#include <hip/hip_runtime.h>
#include <hip/hip_bf16.h>
#include <stdint.h>

// ViG Grapher+FFN block for gfx1250 (MI455X), wave32 + WMMA f32_16x16x32_f16
// + Tensor Data Mover (TDM) double-buffered LDS staging.
// B=16, C=96, H=W=56, N=3136, NR=784, K=9.

typedef _Float16 v16h __attribute__((ext_vector_type(16)));
typedef _Float16 v8h  __attribute__((ext_vector_type(8)));
typedef float    v8f  __attribute__((ext_vector_type(8)));
typedef unsigned int u32x4 __attribute__((ext_vector_type(4)));
typedef int          i32x8 __attribute__((ext_vector_type(8)));
typedef int          i32x4 __attribute__((ext_vector_type(4)));

#define BN_EPS   1e-5f
#define NB       16
#define NC       96
#define NSP      3136
#define NRED     784
#define NROWS    (NB * NSP)      // 50176, divisible by 128
#define KNN      9

__device__ __forceinline__ void wave_lds_fence() {
  __builtin_amdgcn_wave_barrier();
  __asm__ volatile("s_wait_dscnt 0" ::: "memory");
  __builtin_amdgcn_wave_barrier();
}

__device__ __forceinline__ float gelu_exact(float v) {
  return 0.5f * v * (1.0f + erff(v * 0.70710678118654752f));
}

// Contiguous 1-D global->LDS copy via the Tensor Data Mover.
// data_size = 8 bytes; nbytes must be a multiple of 8; addresses 8B-aligned.
// Issue from one wave only; completion tracked with TENSORcnt (in-order).
__device__ __forceinline__ void tdm_load_lds_1d(const void* gsrc,
                                                const void* lds_dst,
                                                uint32_t nbytes) {
  uint64_t ga = (uint64_t)(uintptr_t)gsrc;
  uint32_t elems = nbytes >> 3;                    // 8-byte elements
  uint32_t laddr = (uint32_t)(uintptr_t)lds_dst;   // LDS byte address (addr[31:0])
  u32x4 g0 = {};
  g0[0] = 1u;                                      // count=1 valid user descriptor
  g0[1] = laddr;                                   // lds_addr
  g0[2] = (uint32_t)ga;                            // global_addr[31:0]
  g0[3] = ((uint32_t)(ga >> 32) & 0x01FFFFFFu) | (2u << 30);  // addr[56:32] | type=2
  i32x8 g1 = {};
  g1[0] = (int)(3u << 16);                         // data_size = 8B; no mask/flags
  g1[1] = (int)((elems & 0xFFFFu) << 16);          // tensor_dim0[15:0]  (bits 63:48)
  g1[2] = (int)(((elems >> 16) & 0xFFFFu) | (1u << 16)); // dim0[31:16] | tensor_dim1=1
  g1[3] = (int)((elems & 0xFFFFu) << 16);          // tensor_dim1 hi=0 | tile_dim0
  g1[4] = 1;                                       // tile_dim1=1, tile_dim2=0
  g1[5] = (int)elems;                              // tensor_dim0_stride[31:0]
  g1[6] = 0;                                       // stride0 hi | stride1 lo
  g1[7] = 0;
  i32x4 gz = {};
#if defined(__clang_major__) && __clang_major__ >= 23
  i32x8 gz8 = {};
  __builtin_amdgcn_tensor_load_to_lds(g0, g1, gz, gz, gz8, 0);
#else
  __builtin_amdgcn_tensor_load_to_lds(g0, g1, gz, gz, 0);
#endif
}

// ---------------- prep kernels ----------------
__global__ void kconv_f16(const float* __restrict__ w, _Float16* __restrict__ wh, int n) {
  int i = blockIdx.x * 256 + threadIdx.x;
  if (i < n) wh[i] = (_Float16)w[i];
}

__global__ void kfold(const float* __restrict__ bias, const float* __restrict__ g,
                      const float* __restrict__ bb, const float* __restrict__ m,
                      const float* __restrict__ v, float* __restrict__ s,
                      float* __restrict__ t, int n) {
  int i = blockIdx.x * 256 + threadIdx.x;
  if (i < n) {
    float sc = g[i] * rsqrtf(v[i] + BN_EPS);
    s[i] = sc;
    t[i] = bias[i] * sc + bb[i] - m[i] * sc;
  }
}

// x (B,C,N) f32 -> xh (B,N,C) f16 (tiled transpose)
__global__ void ktrans(const float* __restrict__ x, _Float16* __restrict__ xh) {
  __shared__ float tile[32][33];
  int b = blockIdx.z, n0 = blockIdx.x * 32, c0 = blockIdx.y * 32;
  int tx = threadIdx.x, ty = threadIdx.y;
#pragma unroll
  for (int j = 0; j < 4; j++)
    tile[ty + 8 * j][tx] = x[((size_t)b * NC + c0 + ty + 8 * j) * NSP + n0 + tx];
  __syncthreads();
#pragma unroll
  for (int j = 0; j < 4; j++)
    xh[((size_t)b * NSP + n0 + ty + 8 * j) * NC + c0 + tx] = (_Float16)tile[tx][ty + 8 * j];
}

// ---------------- generic WMMA GEMM + fused epilogue ----------------
// Out[row,o] = epilogue( sum_k A[row,k] * W[o,k] ), rows = B*N (NHWC).
// A-fragments hoisted into registers; 16-row weight slabs double-buffered in LDS
// via TDM, DMA of slab i+1 overlapped with WMMA compute on slab i.
template <int CIN, int COUT>
__global__ __launch_bounds__(256) void kgemm(
    const _Float16* __restrict__ A, const _Float16* __restrict__ Wg,
    const float* __restrict__ scale, const float* __restrict__ shift, int doGelu,
    const float* __restrict__ addNhwc, const float* __restrict__ addChw,
    float* __restrict__ outF32, _Float16* __restrict__ outF16,
    float* __restrict__ outChw) {
  __shared__ __align__(32) _Float16 lw[2][16 * CIN];   // double-buffered weight slabs
  const int tid = threadIdx.x;
  const int w = tid >> 5, l = tid & 31, hs = l >> 4, lr = l & 15;
  const size_t row0 = (size_t)blockIdx.x * 128;
  const _Float16* arow = A + (row0 + (size_t)w * 16 + lr) * CIN;

  constexpr int NK = CIN / 32;
  constexpr int NT = COUT / 16;
  constexpr uint32_t SLAB = 16 * CIN * 2;              // bytes per weight slab

  if (w == 0) tdm_load_lds_1d(Wg, &lw[0][0], SLAB);    // prefetch slab 0

  v16h afrag[NK];
#pragma unroll
  for (int kk = 0; kk < NK; kk++) {
    const _Float16* ap = arow + kk * 32 + hs * 16;
    v8h alo = *(const v8h*)ap;
    v8h ahi = *(const v8h*)(ap + 8);
    afrag[kk] = __builtin_shufflevector(alo, ahi, 0, 1, 2, 3, 4, 5, 6, 7, 8, 9,
                                        10, 11, 12, 13, 14, 15);
  }

  for (int it = 0; it < NT; ++it) {
    if (w == 0) __builtin_amdgcn_s_wait_tensorcnt(0);  // slab `it` landed
    __syncthreads();                                   // publish; prior reads done
    if (w == 0 && it + 1 < NT)
      tdm_load_lds_1d(Wg + (size_t)(it + 1) * 16 * CIN, &lw[(it + 1) & 1][0], SLAB);

    const _Float16* lwb = lw[it & 1];
    v8f acc = {};
#pragma unroll
    for (int kk = 0; kk < NK; kk++) {
      v16h bfr = *(const v16h*)(lwb + lr * CIN + kk * 32 + hs * 16);
      acc = __builtin_amdgcn_wmma_f32_16x16x32_f16(false, afrag[kk], false, bfr,
                                                   (short)0, acc, false, false);
    }

    const int o = it * 16 + lr;
    const float sc = scale[o], sh = shift[o];
#pragma unroll
    for (int r = 0; r < 8; r++) {
      size_t rowg = row0 + (size_t)w * 16 + r + hs * 8;
      float val = acc[r] * sc + sh;
      if (doGelu) val = gelu_exact(val);
      if (addNhwc) val += addNhwc[rowg * COUT + o];
      if (addChw || outChw) {
        size_t bidx = rowg / NSP;
        size_t nidx = rowg - bidx * NSP;
        size_t chw = (bidx * COUT + o) * NSP + nidx;
        if (addChw) val += addChw[chw];
        if (outChw) outChw[chw] = val;
      }
      if (outF32) outF32[rowg * COUT + o] = val;
      if (outF16) outF16[rowg * COUT + o] = (_Float16)val;
    }
  }
}

// ---------------- 2x2 average pool, NHWC ----------------
__global__ void kpool(const float* __restrict__ hx, float* __restrict__ y) {
  size_t i = (size_t)blockIdx.x * 256 + threadIdx.x;
  if (i >= (size_t)NB * NRED * NC) return;
  int c = (int)(i % NC);
  size_t t = i / NC;
  int nr = (int)(t % NRED);
  int b = (int)(t / NRED);
  int hr = nr / 28, wr = nr % 28;
  int n00 = (2 * hr) * 56 + 2 * wr;
  const float* base = hx + ((size_t)b * NSP) * NC + c;
  float v = base[(size_t)n00 * NC] + base[(size_t)(n00 + 1) * NC] +
            base[(size_t)(n00 + 56) * NC] + base[(size_t)(n00 + 57) * NC];
  y[i] = 0.25f * v;
}

// ---------------- per-row L2 normalize (C=96), f32 -> f16 ----------------
__global__ __launch_bounds__(256) void knorm(const float* __restrict__ src,
                                             _Float16* __restrict__ dst, int rows) {
  int w = threadIdx.x >> 5, l = threadIdx.x & 31;
  size_t row = (size_t)blockIdx.x * 8 + w;
  if (row >= (size_t)rows) return;
  const float* p = src + row * NC;
  float v0 = p[l], v1 = p[l + 32], v2 = p[l + 64];
  float s = v0 * v0 + v1 * v1 + v2 * v2;
#pragma unroll
  for (int off = 16; off > 0; off >>= 1) s += __shfl_xor(s, off, 32);
  float inv = 1.0f / fmaxf(sqrtf(s), 1e-12f);
  _Float16* q = dst + row * NC;
  q[l] = (_Float16)(v0 * inv);
  q[l + 32] = (_Float16)(v1 * inv);
  q[l + 64] = (_Float16)(v2 * inv);
}

// ---------------- fused distance GEMM + top-9 selection ----------------
// dist rank key = rel_pos - 2*inner (xsq=ysq=1 after normalization).
// yn chunks (112 rows x 96ch f16 = 21504B) double-buffered in LDS via TDM.
__global__ __launch_bounds__(256) void kdist(const _Float16* __restrict__ xnh,
                                             const _Float16* __restrict__ ynh,
                                             const float* __restrict__ rel,
                                             int* __restrict__ nn) {
  __shared__ __align__(32) _Float16 yl[2][112 * NC];
  __shared__ float db[8][16][17];
  const int tid = threadIdx.x;
  const int w = tid >> 5, l = tid & 31, hs = l >> 4, lr = l & 15;
  const int b = blockIdx.y;
  const int t = blockIdx.x * 8 + w;        // n-tile 0..195 (196 tiles of 16 rows)
  const bool active = (t < NSP / 16);
  const int n0 = t * 16;
  const _Float16* ysrc = ynh + (size_t)b * NRED * NC;
  constexpr uint32_t CHUNK = 112 * NC * 2;

  if (w == 0) tdm_load_lds_1d(ysrc, &yl[0][0], CHUNK);   // prefetch chunk 0

  v16h a0 = {}, a1 = {}, a2 = {};
  if (active) {
    const _Float16* ap = xnh + ((size_t)b * NSP + n0 + lr) * NC + hs * 16;
#pragma unroll
    for (int kk = 0; kk < 3; kk++) {
      v8h alo = *(const v8h*)(ap + kk * 32);
      v8h ahi = *(const v8h*)(ap + kk * 32 + 8);
      v16h afr = __builtin_shufflevector(alo, ahi, 0, 1, 2, 3, 4, 5, 6, 7, 8, 9,
                                         10, 11, 12, 13, 14, 15);
      if (kk == 0) a0 = afr; else if (kk == 1) a1 = afr; else a2 = afr;
    }
  }

  float bv[KNN];
  int bi[KNN];
#pragma unroll
  for (int q = 0; q < KNN; q++) { bv[q] = 3.0e38f; bi[q] = 0; }

  for (int ch = 0; ch < 7; ch++) {         // 7 chunks of 112 nr-rows = 784
    if (w == 0) __builtin_amdgcn_s_wait_tensorcnt(0);    // chunk `ch` landed
    __syncthreads();
    if (w == 0 && ch + 1 < 7)
      tdm_load_lds_1d(ysrc + (size_t)(ch + 1) * 112 * NC, &yl[(ch + 1) & 1][0], CHUNK);
    const _Float16* ylb = yl[ch & 1];
    const int cbase = ch * 112;

    if (active) {
      for (int nt = 0; nt < 7; nt++) {     // 7 nr-tiles of 16 per chunk
        v8f acc = {};
        const _Float16* bl = ylb + (nt * 16 + lr) * NC + hs * 16;
        v16h b0 = *(const v16h*)(bl);
        v16h b1 = *(const v16h*)(bl + 32);
        v16h b2 = *(const v16h*)(bl + 64);
        acc = __builtin_amdgcn_wmma_f32_16x16x32_f16(false, a0, false, b0, (short)0, acc, false, false);
        acc = __builtin_amdgcn_wmma_f32_16x16x32_f16(false, a1, false, b1, (short)0, acc, false, false);
        acc = __builtin_amdgcn_wmma_f32_16x16x32_f16(false, a2, false, b2, (short)0, acc, false, false);
#pragma unroll
        for (int r = 0; r < 8; r++) db[w][r + hs * 8][lr] = acc[r];
        wave_lds_fence();
        if (l < 16) {
          const float* rp = rel + (size_t)(n0 + l) * NRED + cbase + nt * 16;
          const int base_idx = cbase + nt * 16;
#pragma unroll
          for (int j = 0; j < 16; j++) {
            float cand = rp[j] - 2.0f * db[w][l][j];
            if (cand < bv[0]) {            // bv sorted descending, bv[0] = worst
              bv[0] = cand; bi[0] = base_idx + j;
#pragma unroll
              for (int q = 0; q < KNN - 1; q++) {
                if (bv[q] < bv[q + 1]) {
                  float tv = bv[q]; bv[q] = bv[q + 1]; bv[q + 1] = tv;
                  int ti = bi[q]; bi[q] = bi[q + 1]; bi[q + 1] = ti;
                }
              }
            }
          }
        }
        wave_lds_fence();
      }
    }
  }

  if (active && l < 16) {
    int* op = nn + ((size_t)b * NSP + n0 + l) * KNN;
#pragma unroll
    for (int q = 0; q < KNN; q++) op[q] = bi[q];
  }
}

// ---------------- gather + max-relative message + interleaved concat ----------------
__global__ __launch_bounds__(256) void kmsg(const float* __restrict__ hx,
                                            const float* __restrict__ y,
                                            const int* __restrict__ nn,
                                            _Float16* __restrict__ cat) {
  const int w = threadIdx.x >> 5, l = threadIdx.x & 31;
  const size_t row = (size_t)blockIdx.x * 8 + w;
  const int b = (int)(row / NSP);
  const int* ip = nn + row * KNN;
  int id[KNN];
#pragma unroll
  for (int q = 0; q < KNN; q++) id[q] = ip[q];
  const float* hxr = hx + row * NC;
  const float* yb = y + (size_t)b * NRED * NC;
  uint32_t* cr = (uint32_t*)(cat + row * (2 * NC));
#pragma unroll
  for (int ci = 0; ci < 3; ci++) {
    int c = l + ci * 32;
    float xi = hxr[c];
    float m = -3.0e38f;
#pragma unroll
    for (int q = 0; q < KNN; q++) m = fmaxf(m, yb[(size_t)id[q] * NC + c]);
    union { _Float16 h[2]; uint32_t u; } pk;
    pk.h[0] = (_Float16)xi;        // even channel 2c  = x_i
    pk.h[1] = (_Float16)(m - xi);  // odd  channel 2c+1 = msg
    cr[c] = pk.u;
  }
}

// ---------------- host launcher ----------------
extern "C" void kernel_launch(void* const* d_in, const int* in_sizes, int n_in,
                              void* d_out, int out_size, void* d_ws, size_t ws_size,
                              hipStream_t stream) {
  (void)in_sizes; (void)n_in; (void)out_size; (void)ws_size;
  const float* x     = (const float*)d_in[0];
  const float* w1    = (const float*)d_in[1];
  const float* b1    = (const float*)d_in[2];
  const float* bn1g  = (const float*)d_in[3];
  const float* bn1b  = (const float*)d_in[4];
  const float* bn1m  = (const float*)d_in[5];
  const float* bn1v  = (const float*)d_in[6];
  const float* rel   = (const float*)d_in[7];
  const float* gcw   = (const float*)d_in[8];
  const float* gcb   = (const float*)d_in[9];
  const float* gbng  = (const float*)d_in[10];
  const float* gbnb  = (const float*)d_in[11];
  const float* gbnm  = (const float*)d_in[12];
  const float* gbnv  = (const float*)d_in[13];
  const float* w2    = (const float*)d_in[14];
  const float* b2    = (const float*)d_in[15];
  const float* bn2g  = (const float*)d_in[16];
  const float* bn2b  = (const float*)d_in[17];
  const float* bn2m  = (const float*)d_in[18];
  const float* bn2v  = (const float*)d_in[19];
  const float* fw1   = (const float*)d_in[20];
  const float* fb1   = (const float*)d_in[21];
  const float* fbn1g = (const float*)d_in[22];
  const float* fbn1b = (const float*)d_in[23];
  const float* fbn1m = (const float*)d_in[24];
  const float* fbn1v = (const float*)d_in[25];
  const float* fw2   = (const float*)d_in[26];
  const float* fb2   = (const float*)d_in[27];
  const float* fbn2g = (const float*)d_in[28];
  const float* fbn2b = (const float*)d_in[29];
  const float* fbn2m = (const float*)d_in[30];
  const float* fbn2v = (const float*)d_in[31];

  char* ws = (char*)d_ws;
  size_t off = 0;
  auto alloc = [&](size_t bytes) -> void* {
    void* p = ws + off;
    off += (bytes + 255) & ~(size_t)255;
    return p;
  };

  _Float16* wh1   = (_Float16*)alloc((size_t)96 * 96 * 2);
  _Float16* whgc  = (_Float16*)alloc((size_t)192 * 192 * 2);
  _Float16* whf2  = (_Float16*)alloc((size_t)96 * 192 * 2);
  _Float16* whn1  = (_Float16*)alloc((size_t)384 * 96 * 2);
  _Float16* whn2  = (_Float16*)alloc((size_t)96 * 384 * 2);
  float* s1 = (float*)alloc(96 * 4);  float* t1 = (float*)alloc(96 * 4);
  float* s2 = (float*)alloc(192 * 4); float* t2 = (float*)alloc(192 * 4);
  float* s3 = (float*)alloc(96 * 4);  float* t3 = (float*)alloc(96 * 4);
  float* s4 = (float*)alloc(384 * 4); float* t4 = (float*)alloc(384 * 4);
  float* s5 = (float*)alloc(96 * 4);  float* t5 = (float*)alloc(96 * 4);

  _Float16* xh     = (_Float16*)alloc((size_t)NROWS * 96 * 2);   // reused as xn after GEMM1
  float*    hx     = (float*)alloc((size_t)NROWS * 96 * 4);
  float*    yf     = (float*)alloc((size_t)NB * NRED * 96 * 4);
  _Float16* ynh    = (_Float16*)alloc((size_t)NB * NRED * 96 * 2);
  int*      nn     = (int*)alloc((size_t)NROWS * KNN * 4);
  _Float16* cath   = (_Float16*)alloc((size_t)NROWS * 192 * 2);
  _Float16* gh     = (_Float16*)alloc((size_t)NROWS * 192 * 2);
  float*    score  = (float*)alloc((size_t)NROWS * 96 * 4);
  _Float16* scoreh = (_Float16*)alloc((size_t)NROWS * 96 * 2);
  _Float16* ffnh   = (_Float16*)alloc((size_t)NROWS * 384 * 2);

  // prep: weight conversion + BN folding
  kconv_f16<<<(9216 + 255) / 256, 256, 0, stream>>>(w1, wh1, 9216);
  kconv_f16<<<(36864 + 255) / 256, 256, 0, stream>>>(gcw, whgc, 36864);
  kconv_f16<<<(18432 + 255) / 256, 256, 0, stream>>>(w2, whf2, 18432);
  kconv_f16<<<(36864 + 255) / 256, 256, 0, stream>>>(fw1, whn1, 36864);
  kconv_f16<<<(36864 + 255) / 256, 256, 0, stream>>>(fw2, whn2, 36864);
  kfold<<<1, 256, 0, stream>>>(b1, bn1g, bn1b, bn1m, bn1v, s1, t1, 96);
  kfold<<<1, 256, 0, stream>>>(gcb, gbng, gbnb, gbnm, gbnv, s2, t2, 192);
  kfold<<<1, 256, 0, stream>>>(b2, bn2g, bn2b, bn2m, bn2v, s3, t3, 96);
  kfold<<<2, 256, 0, stream>>>(fb1, fbn1g, fbn1b, fbn1m, fbn1v, s4, t4, 384);
  kfold<<<1, 256, 0, stream>>>(fb2, fbn2g, fbn2b, fbn2m, fbn2v, s5, t5, 96);

  // NCHW -> NHWC f16
  ktrans<<<dim3(98, 3, 16), dim3(32, 8), 0, stream>>>(x, xh);

  // Grapher fc1 + BN -> hx (NHWC f32)
  kgemm<96, 96><<<NROWS / 128, 256, 0, stream>>>(xh, wh1, s1, t1, 0, nullptr,
                                                 nullptr, hx, nullptr, nullptr);
  // pool + normalize
  kpool<<<(NB * NRED * 96) / 256, 256, 0, stream>>>(hx, yf);
  knorm<<<NROWS / 8, 256, 0, stream>>>(hx, xh, NROWS);              // xn (reuses xh)
  knorm<<<(NB * NRED) / 8, 256, 0, stream>>>(yf, ynh, NB * NRED);   // yn

  // fused distance GEMM + top-9
  kdist<<<dim3(25, NB), 256, 0, stream>>>(xh, ynh, rel, nn);

  // gather + max message + interleaved concat (f16)
  kmsg<<<NROWS / 8, 256, 0, stream>>>(hx, yf, nn, cath);

  // gc conv + BN + GELU -> gh (f16)
  kgemm<192, 192><<<NROWS / 128, 256, 0, stream>>>(cath, whgc, s2, t2, 1, nullptr,
                                                   nullptr, nullptr, gh, nullptr);
  // fc2 + BN + shortcut(x, CHW) -> score (f32 + f16)
  kgemm<192, 96><<<NROWS / 128, 256, 0, stream>>>(gh, whf2, s3, t3, 0, nullptr,
                                                  x, score, scoreh, nullptr);
  // FFN fc1 + BN + GELU -> ffnh (f16)
  kgemm<96, 384><<<NROWS / 128, 256, 0, stream>>>(scoreh, whn1, s4, t4, 1, nullptr,
                                                  nullptr, nullptr, ffnh, nullptr);
  // FFN fc2 + BN + residual(score) -> d_out (CHW f32)
  kgemm<384, 96><<<NROWS / 128, 256, 0, stream>>>(ffnh, whn2, s5, t5, 0, score,
                                                  nullptr, nullptr, nullptr,
                                                  (float*)d_out);
}
